// EndoLoss_13615046328886
// MI455X (gfx1250) — compile-verified, matrix operand order checked by reading, not run on hardware
//
#include <hip/hip_runtime.h>
#include <hip/hip_bf16.h>

#define N_DIM 32768
#define C_DIM 1000
#define A_DIM 512

typedef __attribute__((ext_vector_type(16))) __bf16 v16bf;
typedef __attribute__((ext_vector_type(8)))  float  v8f;

__device__ __forceinline__ unsigned short f2bf16(float f) {
  unsigned u = __float_as_uint(f);
  u += 0x7FFFu + ((u >> 16) & 1u);               // round-to-nearest-even
  return (unsigned short)(u >> 16);
}
__device__ __forceinline__ unsigned pack2bf(float lo, float hi) {
  return (unsigned)f2bf16(lo) | ((unsigned)f2bf16(hi) << 16);
}

// ---------------- stats kernels (update_CV) ----------------

__global__ void zero_k(float* p, int n) {
  int i = blockIdx.x * blockDim.x + threadIdx.x;
  if (i < n) p[i] = 0.f;
}

__global__ void count_k(const int* __restrict__ labels, float* __restrict__ counts) {
  int i = blockIdx.x * blockDim.x + threadIdx.x;
  if (i < N_DIM) atomicAdd(&counts[labels[i]], 1.0f);
}

__global__ void sumf_k(const float* __restrict__ feat, const int* __restrict__ labels,
                       float* __restrict__ sumF) {
  int i = blockIdx.x;                 // row
  int l = labels[i];
  const float* f = feat + (size_t)i * A_DIM;
  float* s = sumF + (size_t)l * A_DIM;
  int a = threadIdx.x;                // 256 threads, 2 elems each
  atomicAdd(&s[a],        f[a]);
  atomicAdd(&s[a + 256],  f[a + 256]);
}

__global__ void ave_k(const float* __restrict__ sumF, const float* __restrict__ counts,
                      float* __restrict__ ave) {
  int idx = blockIdx.x * blockDim.x + threadIdx.x;
  if (idx < C_DIM * A_DIM) {
    int c = idx >> 9;
    float cnt = counts[c];
    float d = (cnt == 0.f) ? 1.f : cnt;
    ave[idx] = sumF[idx] / d;
  }
}

__global__ void var_k(const float* __restrict__ feat, const int* __restrict__ labels,
                      const float* __restrict__ ave, float* __restrict__ sumVar) {
  int i = blockIdx.x;
  int l = labels[i];
  const float* f  = feat + (size_t)i * A_DIM;
  const float* av = ave  + (size_t)l * A_DIM;
  float* sv = sumVar + (size_t)l * A_DIM;
  int a = threadIdx.x;
  float d0 = f[a] - av[a];
  float d1 = f[a + 256] - av[a + 256];
  atomicAdd(&sv[a],       d0 * d0);
  atomicAdd(&sv[a + 256], d1 * d1);
}

__global__ void cv_k(const float* __restrict__ sumVar, const float* __restrict__ counts,
                     const float* __restrict__ amount_prev, const float* __restrict__ cov_prev,
                     const float* __restrict__ ave_prev, const float* __restrict__ aveArr,
                     float* __restrict__ cvOut) {
  int idx = blockIdx.x * blockDim.x + threadIdx.x;
  if (idx < C_DIM * A_DIM) {
    int c = idx >> 9;
    float cnt = counts[c];
    float denom = (cnt == 0.f) ? 1.f : cnt;
    float var = sumVar[idx] / denom;
    float tot = cnt + amount_prev[c];
    float w = (tot == 0.f) ? 0.f : cnt / tot;
    float dm = ave_prev[idx] - aveArr[idx];
    cvOut[idx] = cov_prev[idx] * (1.f - w) + var * w + w * (1.f - w) * dm * dm;
  }
}

// ---------------- pack B into WMMA bf16 B-operand lane layout ----------------
// B is (K=1024) x (Npad=1024): B[k][j] = k<512 ? W[j][k]^2 : W[j][k-512].
// Tile (kt in 0..31 [K/32], nt in 0..63 [N/16]) stored as 32 lanes x 8 dwords,
// lane L: col j = nt*16 + (L&15), khalf = L>>4, dword d holds K pair
//   k = kt*32 + (d/4)*16 + khalf*8 + (d%4)*2   (matches ISA 16-bit operand layout)

__device__ __forceinline__ float bval(const float* __restrict__ w, int j, int k) {
  if (k < A_DIM) { float x = w[(size_t)j * A_DIM + k]; return x * x; }
  return w[(size_t)j * A_DIM + (k - A_DIM)];
}

__global__ void packb_k(const float* __restrict__ weight, unsigned* __restrict__ b_pack) {
  int idx = blockIdx.x * blockDim.x + threadIdx.x;      // 32*64*32 = 65536
  if (idx >= 32 * 64 * 32) return;
  int lane = idx & 31;
  int nt   = (idx >> 5) & 63;
  int kt   = idx >> 11;
  int j = nt * 16 + (lane & 15);
  int h = lane >> 4;
  unsigned* dst = b_pack + ((size_t)(kt * 64 + nt)) * 256 + lane * 8;
#pragma unroll
  for (int d = 0; d < 8; ++d) {
    int k0 = kt * 32 + (d >> 2) * 16 + h * 8 + (d & 3) * 2;
    float v0 = 0.f, v1 = 0.f;
    if (j < C_DIM) { v0 = bval(weight, j, k0); v1 = bval(weight, j, k0 + 1); }
    dst[d] = pack2bf(v0, v1);
  }
}

// ---------------- t3[i] = sum_a cv[y_i,a] * W[y_i,a]^2 (one wave per row) ----------------

__global__ void t3_k(const float* __restrict__ cv, const float* __restrict__ weight,
                     const int* __restrict__ labels, float* __restrict__ t3) {
  int row  = blockIdx.x * 8 + (threadIdx.x >> 5);
  int lane = threadIdx.x & 31;
  if (row >= N_DIM) return;
  int l = labels[row];
  const float* c = cv + (size_t)l * A_DIM;
  const float* w = weight + (size_t)l * A_DIM;
  float s = 0.f;
  for (int a = lane; a < A_DIM; a += 32) { float wv = w[a]; s += c[a] * wv * wv; }
#pragma unroll
  for (int off = 1; off < 32; off <<= 1) s += __shfl_xor(s, off, 32);
  if (lane == 0) t3[row] = s;
}

// ---------------- fused GEMM (bf16 WMMA) + logits add + log-softmax + NLL ----------------
// Grid: N/32 blocks x 512 threads (16 waves). Block owns 32 rows x 1024 cols.
// Wave (mlocal, nw): M-tile mlocal of 2, column strip nw of 8 (128 cols = 8 tiles).
// A operand (32 rows x K=1024 bf16, WMMA lane layout) is staged once in LDS by all
// 16 waves cooperatively; the K-loop then runs ds_load_b128 (A) + pipelined
// global_load_b128 (B, L2-resident packed layout) + v_wmma per tile.

__global__ __launch_bounds__(512)
void gemm_softmax_k(const float* __restrict__ cv, const float* __restrict__ weight,
                    const unsigned* __restrict__ b_pack, const float* __restrict__ logits,
                    const int* __restrict__ labels, const float* __restrict__ t3,
                    const int* __restrict__ ratio_p, float* __restrict__ out) {
  __shared__ unsigned apack[2][32][256];   // [mtile-local][kt][lane*8 dwords] = 64 KB
  __shared__ float lmax[2][16][8];
  __shared__ float lsum[2][16][8];
  __shared__ float blockLoss;

  const int tid    = threadIdx.x;
  const int lane   = tid & 31;
  const int w      = tid >> 5;       // 0..15
  const int mlocal = w >> 3;         // 0..1
  const int nw     = w & 7;          // 0..7
  const int mtile  = blockIdx.x * 2 + mlocal;
  const int mbase  = mtile * 16;
  const int m      = lane & 15;
  const int h      = lane >> 4;
  const int cidx   = lane & 15;
  if (tid == 0) blockLoss = 0.f;

  // ---- cooperative A-operand staging: 64 chunks (2 mtiles x 32 kt), 4 per wave ----
#pragma unroll
  for (int j = 0; j < 4; ++j) {
    const int c  = w * 4 + j;        // 0..63
    const int ml = c >> 5;
    const int kt = c & 31;
    const int mt = blockIdx.x * 2 + ml;
    const int lr = labels[mt * 16 + m];
    const float* cr  = cv + (size_t)lr * A_DIM;
    const float* wrr = weight + (size_t)lr * A_DIM;
    float va[16];
#pragma unroll
    for (int g = 0; g < 2; ++g) {
      const int k0 = kt * 32 + g * 16 + h * 8;
      if (k0 < A_DIM) {              // K < 512 -> cv
        const float4 c0 = *(const float4*)(cr + k0);
        const float4 c1 = *(const float4*)(cr + k0 + 4);
        va[g*8+0]=c0.x; va[g*8+1]=c0.y; va[g*8+2]=c0.z; va[g*8+3]=c0.w;
        va[g*8+4]=c1.x; va[g*8+5]=c1.y; va[g*8+6]=c1.z; va[g*8+7]=c1.w;
      } else {                       // K >= 512 -> -2*cv*W
        const int a0 = k0 - A_DIM;
        const float4 c0 = *(const float4*)(cr + a0);
        const float4 c1 = *(const float4*)(cr + a0 + 4);
        const float4 w0 = *(const float4*)(wrr + a0);
        const float4 w1 = *(const float4*)(wrr + a0 + 4);
        va[g*8+0]=-2.f*c0.x*w0.x; va[g*8+1]=-2.f*c0.y*w0.y;
        va[g*8+2]=-2.f*c0.z*w0.z; va[g*8+3]=-2.f*c0.w*w0.w;
        va[g*8+4]=-2.f*c1.x*w1.x; va[g*8+5]=-2.f*c1.y*w1.y;
        va[g*8+6]=-2.f*c1.z*w1.z; va[g*8+7]=-2.f*c1.w*w1.w;
      }
    }
    uint4 q0, q1;
    q0.x = pack2bf(va[0],  va[1]);  q0.y = pack2bf(va[2],  va[3]);
    q0.z = pack2bf(va[4],  va[5]);  q0.w = pack2bf(va[6],  va[7]);
    q1.x = pack2bf(va[8],  va[9]);  q1.y = pack2bf(va[10], va[11]);
    q1.z = pack2bf(va[12], va[13]); q1.w = pack2bf(va[14], va[15]);
    unsigned* dst = &apack[ml][kt][lane * 8];
    *(uint4*)dst       = q0;
    *(uint4*)(dst + 4) = q1;
  }
  __syncthreads();

  v8f acc[8];
  const v8f vzero = {0.f, 0.f, 0.f, 0.f, 0.f, 0.f, 0.f, 0.f};
#pragma unroll
  for (int t = 0; t < 8; ++t) acc[t] = vzero;

#pragma unroll 1
  for (int kt = 0; kt < 32; ++kt) {
    // A fragment from LDS (2x ds_load_b128)
    union { uint4 q[2]; v16bf v; } afr;
    const unsigned* asrc = &apack[mlocal][kt][lane * 8];
    afr.q[0] = *(const uint4*)asrc;
    afr.q[1] = *(const uint4*)(asrc + 4);

    // B fragments: software-pipelined, tile stride = 64 dwords = 16 uint4
    const uint4* bp = (const uint4*)(b_pack + (size_t)kt * 16384 +
                                     (size_t)(nw * 8) * 256) + lane * 2;
    __builtin_prefetch((const void*)(bp + 4096), 0, 0);   // next kt slice
    union BU { uint4 q[2]; v16bf v; } bbuf[2];
    bbuf[0].q[0] = bp[0];
    bbuf[0].q[1] = bp[1];
#pragma unroll
    for (int t = 0; t < 8; ++t) {
      if (t < 7) {
        bbuf[(t + 1) & 1].q[0] = bp[(t + 1) * 16];
        bbuf[(t + 1) & 1].q[1] = bp[(t + 1) * 16 + 1];
      }
      acc[t] = __builtin_amdgcn_wmma_f32_16x16x32_bf16(
          false, afr.v, false, bbuf[t & 1].v, (short)0, acc[t], false, false);
    }
  }

  // ---- epilogue: aug = logits + 0.5*ratio*(acc + t3[row]); padded cols -> -inf ----
  const float coef = 0.5f * (float)(ratio_p[0]);
  float t3v[8];
  int   labv[8];
#pragma unroll
  for (int r = 0; r < 8; ++r) {
    const int row = mbase + h * 8 + r;
    t3v[r]  = t3[row];
    labv[r] = labels[row];
  }
#pragma unroll
  for (int t = 0; t < 8; ++t) {
    const int col = (nw * 8 + t) * 16 + cidx;
#pragma unroll
    for (int r = 0; r < 8; ++r) {
      const int row = mbase + h * 8 + r;
      float v;
      if (col < C_DIM) {
        const float lg = logits[(size_t)row * C_DIM + col];
        v = lg + coef * (acc[t][r] + t3v[r]);
      } else {
        v = -__builtin_inff();
      }
      acc[t][r] = v;
    }
  }

  // row max: reduce over tiles, then half-wave butterfly, then cross-wave via LDS
  float rm[8];
#pragma unroll
  for (int r = 0; r < 8; ++r) {
    float mv = acc[0][r];
#pragma unroll
    for (int t = 1; t < 8; ++t) mv = fmaxf(mv, acc[t][r]);
#pragma unroll
    for (int off = 1; off < 16; off <<= 1) mv = fmaxf(mv, __shfl_xor(mv, off, 32));
    rm[r] = mv;
  }
  if (cidx == 0) {
#pragma unroll
    for (int r = 0; r < 8; ++r) lmax[mlocal][h * 8 + r][nw] = rm[r];
  }
  __syncthreads();
#pragma unroll
  for (int r = 0; r < 8; ++r) {
    float mv = -__builtin_inff();
#pragma unroll
    for (int q = 0; q < 8; ++q) mv = fmaxf(mv, lmax[mlocal][h * 8 + r][q]);
    rm[r] = mv;
  }

  // sum of exp
  float se[8];
#pragma unroll
  for (int r = 0; r < 8; ++r) {
    float s = 0.f;
#pragma unroll
    for (int t = 0; t < 8; ++t) s += __expf(acc[t][r] - rm[r]);
#pragma unroll
    for (int off = 1; off < 16; off <<= 1) s += __shfl_xor(s, off, 32);
    se[r] = s;
  }
  if (cidx == 0) {
#pragma unroll
    for (int r = 0; r < 8; ++r) lsum[mlocal][h * 8 + r][nw] = se[r];
  }
  __syncthreads();
#pragma unroll
  for (int r = 0; r < 8; ++r) {
    float s = 0.f;
#pragma unroll
    for (int q = 0; q < 8; ++q) s += lsum[mlocal][h * 8 + r][q];
    se[r] = s;
  }

  // NLL at target column: logZ - aug[row, label]
  float contrib = 0.f;
  bool any = false;
#pragma unroll
  for (int t = 0; t < 8; ++t) {
    const int col = (nw * 8 + t) * 16 + cidx;
#pragma unroll
    for (int r = 0; r < 8; ++r) {
      if (col == labv[r]) {
        contrib += rm[r] + __logf(se[r]) - acc[t][r];
        any = true;
      }
    }
  }
  if (any) atomicAdd(&blockLoss, contrib);
  __syncthreads();
  if (tid == 0) atomicAdd(out, blockLoss * (1.0f / (float)N_DIM));
}

// ---------------- launch ----------------

extern "C" void kernel_launch(void* const* d_in, const int* in_sizes, int n_in,
                              void* d_out, int out_size, void* d_ws, size_t ws_size,
                              hipStream_t stream) {
  const float* weight      = (const float*)d_in[0];
  const float* logits      = (const float*)d_in[1];
  const float* features    = (const float*)d_in[2];
  const int*   labels      = (const int*)d_in[3];
  const float* cov_prev    = (const float*)d_in[4];
  const float* ave_prev    = (const float*)d_in[5];
  const float* amount_prev = (const float*)d_in[6];
  const int*   ratio       = (const int*)d_in[7];
  float* out = (float*)d_out;

  // workspace layout (dwords)
  float* ws = (float*)d_ws;
  float*    counts = ws;                         // 1024
  float*    sumF   = ws + 1024;                  // 512000
  float*    sumVar = ws + 513024;                // 512000
  float*    ave    = ws + 1025024;               // 512000
  float*    cvbuf  = ws + 1537024;               // 512000
  float*    t3     = ws + 2049024;               // 32768
  unsigned* b_pack = (unsigned*)(ws + 2081792);  // 524288 dwords (2 MB)

  // 1. zero accumulators + output
  zero_k<<<(1025024 + 255) / 256, 256, 0, stream>>>(ws, 1025024);
  zero_k<<<1, 32, 0, stream>>>(out, 1);
  // 2. class counts
  count_k<<<N_DIM / 256, 256, 0, stream>>>(labels, counts);
  // 3. per-class feature sums
  sumf_k<<<N_DIM, 256, 0, stream>>>(features, labels, sumF);
  // 4. means
  ave_k<<<(C_DIM * A_DIM + 255) / 256, 256, 0, stream>>>(sumF, counts, ave);
  // 5. per-class squared deviations
  var_k<<<N_DIM, 256, 0, stream>>>(features, labels, ave, sumVar);
  // 6. blended covariance
  cv_k<<<(C_DIM * A_DIM + 255) / 256, 256, 0, stream>>>(sumVar, counts, amount_prev,
                                                        cov_prev, ave_prev, ave, cvbuf);
  // 7. pack B operand (WMMA bf16 lane layout, zero-padded C->1024)
  packb_k<<<(32 * 64 * 32) / 256, 256, 0, stream>>>(weight, b_pack);
  // 8. per-row t3
  t3_k<<<N_DIM / 8, 256, 0, stream>>>(cvbuf, weight, labels, t3);
  // 9. fused GEMM + softmax + NLL
  gemm_softmax_k<<<N_DIM / 32, 512, 0, stream>>>(cvbuf, weight, b_pack, logits,
                                                 labels, t3, ratio, out);
}
